// DFIL_21260088115627
// MI455X (gfx1250) — matrix operand order repaired and checked by reading, not compiled, not run on hardware
//
#include <hip/hip_runtime.h>
#include <hip/hip_bf16.h>
#include <math.h>

typedef __attribute__((ext_vector_type(16))) _Float16 v16h;
typedef __attribute__((ext_vector_type(8)))  _Float16 v8h;
typedef __attribute__((ext_vector_type(8)))  float    v8f;

#define CH        256
#define NPTS      4096
#define KNN_K     16
#define ROWS      32768          // B*N = 8*4096
#define MTILES    2048           // ROWS/16
#define GEMM_BLKS 256            // MTILES / 8 waves
#define WPITCH    264            // LDS row pitch in halfs: 528B = 16 mod 256 -> bank-conflict-free
#define SMBYTES   (256 * WPITCH * 2)   // 135168 B per staged weight

// ---------------------------------------------------------------------------
// Async weight staging: global f16 [256][256] -> LDS [256][WPITCH]
// CDNA5 async global->LDS copies (ASYNCcnt) + s_wait_asynccnt.
// ---------------------------------------------------------------------------

__device__ __forceinline__ void stage_issue(const _Float16* __restrict__ w,
                                            _Float16* sm, int tid) {
    int lane = tid & 31;
    int wave = tid >> 5;
    // Each wave copies 32 rows; one b128 per lane covers a 512B row per wave-inst.
#pragma unroll 4
    for (int r = wave; r < 256; r += 8) {
        unsigned long long g = (unsigned long long)(uintptr_t)(w + (size_t)r * CH)
                             + (unsigned long long)(lane * 16);
        unsigned int l = (unsigned int)(uintptr_t)(sm + r * WPITCH) + lane * 16;
        asm volatile("global_load_async_to_lds_b128 %0, %1, off"
                     :: "v"(l), "v"(g) : "memory");
    }
}

__device__ __forceinline__ void stage_wait() {
    asm volatile("s_wait_asynccnt 0" ::: "memory");
    __syncthreads();
}

// ---------------------------------------------------------------------------
// Fragment loaders (CDNA5 WMMA 16x16x32 f16 layouts, ISA 7.12.2)
// ---------------------------------------------------------------------------

// A-matrix 16x32 f16: lane L<16 -> row M=L, K = {0..7, 16..23};
//                     lane L>=16 -> row M=L-16, K = {8..15, 24..31}.
__device__ __forceinline__ v16h afrag_from_f32(const float* __restrict__ rowbase,
                                               int k0, int lane) {
    int r    = lane & 15;
    int koff = (lane >> 4) << 3;                 // 0 or 8
    const float* p = rowbase + (size_t)r * CH + k0 + koff;
    const float4* q = (const float4*)p;
    float4 a0 = q[0], a1 = q[1];                 // K = koff .. koff+7
    float4 b0 = q[4], b1 = q[5];                 // K = koff+16 .. koff+23
    v16h f;
    f[0]=(_Float16)a0.x; f[1]=(_Float16)a0.y; f[2]=(_Float16)a0.z; f[3]=(_Float16)a0.w;
    f[4]=(_Float16)a1.x; f[5]=(_Float16)a1.y; f[6]=(_Float16)a1.z; f[7]=(_Float16)a1.w;
    f[8]=(_Float16)b0.x; f[9]=(_Float16)b0.y; f[10]=(_Float16)b0.z; f[11]=(_Float16)b0.w;
    f[12]=(_Float16)b1.x; f[13]=(_Float16)b1.y; f[14]=(_Float16)b1.z; f[15]=(_Float16)b1.w;
    return f;
}

__device__ __forceinline__ v16h afrag_from_f16(const _Float16* __restrict__ rowbase,
                                               int k0, int lane) {
    int r    = lane & 15;
    int koff = (lane >> 4) << 3;
    const _Float16* p = rowbase + (size_t)r * CH + k0 + koff;
    v8h lo = *(const v8h*)p;
    v8h hi = *(const v8h*)(p + 16);
    v16h f;
#pragma unroll
    for (int i = 0; i < 8; ++i) { f[i] = lo[i]; f[i + 8] = hi[i]; }
    return f;
}

// B-matrix 32x16 f16 (K x N) from LDS-staged weight image:
// lane L<16 -> col N=L (weight row o), K = {0..7, 16..23}; lane L>=16 -> N=L-16, K = {8..15, 24..31}.
__device__ __forceinline__ v16h bfrag_lds(const _Float16* sm, int otile, int k0,
                                          int lane) {
    int o  = (otile << 4) + (lane & 15);
    int kl = (lane >> 4) << 3;                   // 0 or 8
    const _Float16* p = sm + o * WPITCH + k0 + kl;
    v8h lo = *(const v8h*)p;                     // ds_load_b128
    v8h hi = *(const v8h*)(p + 16);
    v16h f;
#pragma unroll
    for (int i = 0; i < 8; ++i) { f[i] = lo[i]; f[i + 8] = hi[i]; }
    return f;
}

// Two 16x16 output tiles over full K=256: two independent WMMA chains.
__device__ __forceinline__ void tile_pair(const v16h* __restrict__ afr,
                                          const _Float16* sm,
                                          int ot, int lane, v8f& c0, v8f& c1) {
    c0 = (v8f){}; c1 = (v8f){};
#pragma unroll
    for (int ks = 0; ks < 8; ++ks) {
        v16h b0 = bfrag_lds(sm, ot,     ks * 32, lane);
        v16h b1 = bfrag_lds(sm, ot + 1, ks * 32, lane);
        c0 = __builtin_amdgcn_wmma_f32_16x16x32_f16(false, afr[ks], false, b0,
                                                    (short)0, c0, false, false);
        c1 = __builtin_amdgcn_wmma_f32_16x16x32_f16(false, afr[ks], false, b1,
                                                    (short)0, c1, false, false);
    }
}

// ---------------------------------------------------------------------------
// Prep kernels
// ---------------------------------------------------------------------------

__global__ void k_cvt_w(const float* __restrict__ src, _Float16* __restrict__ dst, int n) {
    int i = blockIdx.x * blockDim.x + threadIdx.x;
    if (i < n) dst[i] = (_Float16)src[i];
}

// p: [4,256] = gamma,beta,mean,var.  out: [0..255]=scale, [256..511]=shift.
__global__ void k_bn_prep(const float* __restrict__ p, const float* __restrict__ bias,
                          float* __restrict__ out) {
    int c = threadIdx.x;
    float g = p[c], b = p[CH + c], m = p[2 * CH + c], v = p[3 * CH + c];
    float s  = g * rsqrtf(v + 1e-5f);
    float bi = bias ? bias[c] : 0.0f;
    out[c]      = s;
    out[CH + c] = (bi - m) * s + b;
}

// ---------------------------------------------------------------------------
// Stage 1: proj = x @ projW^T (f16 out); x2 = BN_glob(x @ globW^T) (f32 out)
// Single LDS buffer reused for both weights; A fragments stay live in VGPRs.
// ---------------------------------------------------------------------------

__global__ void __launch_bounds__(256) k_proj_glob(
    const float* __restrict__ x,
    const _Float16* __restrict__ wProj, const _Float16* __restrict__ wGlob,
    const float* __restrict__ bnG,
    _Float16* __restrict__ proj16, float* __restrict__ x2) {
    extern __shared__ _Float16 smw[];
    int tid   = threadIdx.x;
    int lane  = tid & 31;
    int wave  = tid >> 5;
    int mtile = blockIdx.x * 8 + wave;
    const float* arow = x + (size_t)mtile * 16 * CH;

    stage_issue(wProj, smw, tid);                 // async copies in flight...

    v16h afr[8];
#pragma unroll
    for (int ks = 0; ks < 8; ++ks) afr[ks] = afrag_from_f32(arow, ks * 32, lane);

    stage_wait();                                 // ...drained + barrier

    int n    = lane & 15;
    int mrow = (lane >> 4) << 3;

    // Pass 1: proj (no BN), f16 out
    for (int ot = 0; ot < 16; ot += 2) {
        v8f c0, c1;
        tile_pair(afr, smw, ot, lane, c0, c1);
#pragma unroll
        for (int half = 0; half < 2; ++half) {
            int o = ((ot + half) << 4) + n;
            size_t idx = ((size_t)(mtile * 16 + mrow)) * CH + o;
#pragma unroll
            for (int r = 0; r < 8; ++r) {
                proj16[idx] = (_Float16)(half ? c1[r] : c0[r]);
                idx += CH;
            }
        }
    }

    __syncthreads();                              // all waves done reading wProj image
    stage_issue(wGlob, smw, tid);
    stage_wait();

    // Pass 2: x2 = BN_glob(x @ globW^T), f32 out
    for (int ot = 0; ot < 16; ot += 2) {
        v8f c0, c1;
        tile_pair(afr, smw, ot, lane, c0, c1);
#pragma unroll
        for (int half = 0; half < 2; ++half) {
            int o = ((ot + half) << 4) + n;
            float s = bnG[o], t = bnG[CH + o];
            size_t idx = ((size_t)(mtile * 16 + mrow)) * CH + o;
#pragma unroll
            for (int r = 0; r < 8; ++r) {
                x2[idx] = (half ? c1[r] : c0[r]) * s + t;
                idx += CH;
            }
        }
    }
}

// ---------------------------------------------------------------------------
// Stage 2: edge maxpool  x_knn[i,c] = max_k( proj[knn[i,k],c] - proj[i,c] )
// ---------------------------------------------------------------------------

__global__ void __launch_bounds__(256) k_edge_max(
    const _Float16* __restrict__ proj16, const long long* __restrict__ knn,
    _Float16* __restrict__ xknn) {
    int row = blockIdx.x;
    int c   = threadIdx.x;
    long long bbase = (long long)(row >> 12) * NPTS;   // batch offset (N=4096)
    float fi = (float)proj16[(size_t)row * CH + c];
    const long long* kn = knn + (size_t)row * KNN_K;
    float m = -3.0e38f;
#pragma unroll
    for (int k = 0; k < KNN_K; ++k) {
        long long nb = bbase + kn[k];
        float fj = (float)proj16[(size_t)nb * CH + c];
        m = fmaxf(m, fj - fi);
    }
    xknn[(size_t)row * CH + c] = (_Float16)m;
}

// ---------------------------------------------------------------------------
// Stage 3: x1 = BN_local(x_knn @ localW^T) (f32 out); f = x1 + x2 (f16 out)
// ---------------------------------------------------------------------------

__global__ void __launch_bounds__(256) k_local(
    const _Float16* __restrict__ xknn, const _Float16* __restrict__ wLocal,
    const float* __restrict__ bnL, const float* __restrict__ x2,
    float* __restrict__ x1, _Float16* __restrict__ fbuf) {
    extern __shared__ _Float16 smw[];
    int tid   = threadIdx.x;
    int lane  = tid & 31;
    int wave  = tid >> 5;
    int mtile = blockIdx.x * 8 + wave;
    const _Float16* arow = xknn + (size_t)mtile * 16 * CH;

    stage_issue(wLocal, smw, tid);
    v16h afr[8];
#pragma unroll
    for (int ks = 0; ks < 8; ++ks) afr[ks] = afrag_from_f16(arow, ks * 32, lane);
    stage_wait();

    int n    = lane & 15;
    int mrow = (lane >> 4) << 3;
    for (int ot = 0; ot < 16; ot += 2) {
        v8f c0, c1;
        tile_pair(afr, smw, ot, lane, c0, c1);
#pragma unroll
        for (int half = 0; half < 2; ++half) {
            int o = ((ot + half) << 4) + n;
            float s = bnL[o], t = bnL[CH + o];
            size_t idx = ((size_t)(mtile * 16 + mrow)) * CH + o;
#pragma unroll
            for (int r = 0; r < 8; ++r) {
                float y = (half ? c1[r] : c0[r]) * s + t;
                x1[idx]   = y;
                fbuf[idx] = (_Float16)(y + x2[idx]);
                idx += CH;
            }
        }
    }
}

// ---------------------------------------------------------------------------
// Stage 4: h = BN_aff1(f @ affW1^T + b1) (f16 out)
// ---------------------------------------------------------------------------

__global__ void __launch_bounds__(256) k_aff1(
    const _Float16* __restrict__ fbuf, const _Float16* __restrict__ wA1,
    const float* __restrict__ bn1, _Float16* __restrict__ h16) {
    extern __shared__ _Float16 smw[];
    int tid   = threadIdx.x;
    int lane  = tid & 31;
    int wave  = tid >> 5;
    int mtile = blockIdx.x * 8 + wave;
    const _Float16* arow = fbuf + (size_t)mtile * 16 * CH;

    stage_issue(wA1, smw, tid);
    v16h afr[8];
#pragma unroll
    for (int ks = 0; ks < 8; ++ks) afr[ks] = afrag_from_f16(arow, ks * 32, lane);
    stage_wait();

    int n    = lane & 15;
    int mrow = (lane >> 4) << 3;
    for (int ot = 0; ot < 16; ot += 2) {
        v8f c0, c1;
        tile_pair(afr, smw, ot, lane, c0, c1);
#pragma unroll
        for (int half = 0; half < 2; ++half) {
            int o = ((ot + half) << 4) + n;
            float s = bn1[o], t = bn1[CH + o];          // bias folded into t
            size_t idx = ((size_t)(mtile * 16 + mrow)) * CH + o;
#pragma unroll
            for (int r = 0; r < 8; ++r) {
                h16[idx] = (_Float16)((half ? c1[r] : c0[r]) * s + t);
                idx += CH;
            }
        }
    }
}

// ---------------------------------------------------------------------------
// Stage 5: a = sigmoid(BN_aff2(h @ affW2^T + b2));
//          out = BN_final(a*x1 + (1-a)*x2)  (f32 out)
// ---------------------------------------------------------------------------

__global__ void __launch_bounds__(256) k_aff2_final(
    const _Float16* __restrict__ h16, const _Float16* __restrict__ wA2,
    const float* __restrict__ bn2, const float* __restrict__ bnF,
    const float* __restrict__ x1, const float* __restrict__ x2,
    float* __restrict__ out) {
    extern __shared__ _Float16 smw[];
    int tid   = threadIdx.x;
    int lane  = tid & 31;
    int wave  = tid >> 5;
    int mtile = blockIdx.x * 8 + wave;
    const _Float16* arow = h16 + (size_t)mtile * 16 * CH;

    stage_issue(wA2, smw, tid);
    v16h afr[8];
#pragma unroll
    for (int ks = 0; ks < 8; ++ks) afr[ks] = afrag_from_f16(arow, ks * 32, lane);
    stage_wait();

    int n    = lane & 15;
    int mrow = (lane >> 4) << 3;
    for (int ot = 0; ot < 16; ot += 2) {
        v8f c0, c1;
        tile_pair(afr, smw, ot, lane, c0, c1);
#pragma unroll
        for (int half = 0; half < 2; ++half) {
            int o = ((ot + half) << 4) + n;
            float s2 = bn2[o], t2 = bn2[CH + o];        // b2 folded into t2
            float sf = bnF[o], tf = bnF[CH + o];
            size_t idx = ((size_t)(mtile * 16 + mrow)) * CH + o;
#pragma unroll
            for (int r = 0; r < 8; ++r) {
                float y = (half ? c1[r] : c0[r]) * s2 + t2;
                float a = 1.0f / (1.0f + expf(-y));
                float fused = a * x1[idx] + (1.0f - a) * x2[idx];
                out[idx] = fused * sf + tf;
                idx += CH;
            }
        }
    }
}

// ---------------------------------------------------------------------------
// Host launcher
// ---------------------------------------------------------------------------

extern "C" void kernel_launch(void* const* d_in, const int* in_sizes, int n_in,
                              void* d_out, int out_size, void* d_ws, size_t ws_size,
                              hipStream_t stream) {
    const float*     x       = (const float*)d_in[0];
    const long long* knn     = (const long long*)d_in[1];
    const float*     projW   = (const float*)d_in[2];
    const float*     localW  = (const float*)d_in[3];
    const float*     globW   = (const float*)d_in[4];
    const float*     affW1   = (const float*)d_in[5];
    const float*     affb1   = (const float*)d_in[6];
    const float*     affW2   = (const float*)d_in[7];
    const float*     affb2   = (const float*)d_in[8];
    const float*     bnLocal = (const float*)d_in[9];
    const float*     bnGlob  = (const float*)d_in[10];
    const float*     bnAff1  = (const float*)d_in[11];
    const float*     bnAff2  = (const float*)d_in[12];
    const float*     bnFinal = (const float*)d_in[13];

    char* ws = (char*)d_ws;
    size_t off = 0;
    auto alloc = [&](size_t bytes) -> char* {
        char* p = ws + off;
        off += (bytes + 255) & ~(size_t)255;
        return p;
    };

    const size_t WELEM = (size_t)CH * CH;            // 65536
    _Float16* wP  = (_Float16*)alloc(WELEM * 2);
    _Float16* wL  = (_Float16*)alloc(WELEM * 2);
    _Float16* wG  = (_Float16*)alloc(WELEM * 2);
    _Float16* wA1 = (_Float16*)alloc(WELEM * 2);
    _Float16* wA2 = (_Float16*)alloc(WELEM * 2);
    float* bnL = (float*)alloc(2 * CH * 4);
    float* bnG = (float*)alloc(2 * CH * 4);
    float* bn1 = (float*)alloc(2 * CH * 4);
    float* bn2 = (float*)alloc(2 * CH * 4);
    float* bnF = (float*)alloc(2 * CH * 4);
    _Float16* proj16 = (_Float16*)alloc((size_t)ROWS * CH * 2);
    float*    x2     = (float*)   alloc((size_t)ROWS * CH * 4);
    _Float16* xknn   = (_Float16*)alloc((size_t)ROWS * CH * 2);
    float*    x1     = (float*)   alloc((size_t)ROWS * CH * 4);
    _Float16* fbuf   = (_Float16*)alloc((size_t)ROWS * CH * 2);
    _Float16* h16    = (_Float16*)alloc((size_t)ROWS * CH * 2);

    // Prep: weights -> f16, BN -> scale/shift (bias folded where present)
    k_cvt_w<<<256, 256, 0, stream>>>(projW,  wP,  (int)WELEM);
    k_cvt_w<<<256, 256, 0, stream>>>(localW, wL,  (int)WELEM);
    k_cvt_w<<<256, 256, 0, stream>>>(globW,  wG,  (int)WELEM);
    k_cvt_w<<<256, 256, 0, stream>>>(affW1,  wA1, (int)WELEM);
    k_cvt_w<<<256, 256, 0, stream>>>(affW2,  wA2, (int)WELEM);
    k_bn_prep<<<1, CH, 0, stream>>>(bnLocal, nullptr, bnL);
    k_bn_prep<<<1, CH, 0, stream>>>(bnGlob,  nullptr, bnG);
    k_bn_prep<<<1, CH, 0, stream>>>(bnAff1,  affb1,   bn1);
    k_bn_prep<<<1, CH, 0, stream>>>(bnAff2,  affb2,   bn2);
    k_bn_prep<<<1, CH, 0, stream>>>(bnFinal, nullptr, bnF);

    // Stage 1: dual GEMM sharing A (LDS-staged weights, two passes)
    k_proj_glob<<<GEMM_BLKS, 256, SMBYTES, stream>>>(x, wP, wG, bnG, proj16, x2);
    // Stage 2: KNN gather + edge max
    k_edge_max<<<ROWS, 256, 0, stream>>>(proj16, knn, xknn);
    // Stage 3: local branch GEMM + BN, f = x1 + x2
    k_local<<<GEMM_BLKS, 256, SMBYTES, stream>>>(xknn, wL, bnL, x2, x1, fbuf);
    // Stage 4: affinity GEMM 1
    k_aff1<<<GEMM_BLKS, 256, SMBYTES, stream>>>(fbuf, wA1, bn1, h16);
    // Stage 5: affinity GEMM 2 + sigmoid blend + final BN
    k_aff2_final<<<GEMM_BLKS, 256, SMBYTES, stream>>>(h16, wA2, bn2, bnF, x1, x2,
                                                      (float*)d_out);
}
